// CPImplicit_23252952941223
// MI455X (gfx1250) — matrix operand order repaired and checked by reading, not compiled, not run on hardware
//
#include <hip/hip_runtime.h>

// ---------------------------------------------------------------------------
// Types / WMMA helpers (gfx1250 wave32 WMMA)
// ---------------------------------------------------------------------------
typedef __attribute__((ext_vector_type(16))) _Float16 v16h;
typedef __attribute__((ext_vector_type(8)))  float    v8f;
typedef unsigned __attribute__((ext_vector_type(4)))  uint4v;
typedef int      __attribute__((ext_vector_type(8)))  int8v;
typedef int      __attribute__((ext_vector_type(4)))  int4v;

__device__ __forceinline__ v16h ldfrag(const _Float16* p0, const _Float16* p1) {
    union { v16h v; uint4 q[2]; } f;
    f.q[0] = *(const uint4*)p0;
    f.q[1] = *(const uint4*)p1;
    return f.v;
}

__device__ __forceinline__ v8f wmma_f16(v16h a, v16h b, v8f c) {
    // D = A(16x32 f16) * B(32x16 f16) + C(16x16 f32)
    return __builtin_amdgcn_wmma_f32_16x16x32_f16(
        /*neg_a=*/false, a, /*neg_b=*/false, b,
        /*c_mod=*/(short)0, c, /*reuse_a=*/false, /*reuse_b=*/false);
}

#define BS    2
#define CCH   64
#define FLAT  16384
#define NPTS  2048
#define KSMP  256
#define EPSB  1e-5f
// ln(10000)/64
#define PE_LOGC 0.14391157f

// ---------------------------------------------------------------------------
// K0: fold resize weights, transpose to f16, precompute BN affine constants
//   sc layout: [0:32)=s1 [32:64)=c1 [64:96)=s2 [96:128)=c2 [128:160)=sne [160:192)=cne
// ---------------------------------------------------------------------------
__global__ __launch_bounds__(256) void k0_prep(
    const float* __restrict__ w_r1, const float* __restrict__ b_r1,
    const float* __restrict__ w_r2, const float* __restrict__ b_r2,
    const float* __restrict__ w_p2,
    const float* __restrict__ b_p1, const float* __restrict__ g_p1, const float* __restrict__ be_p1,
    const float* __restrict__ b_p2, const float* __restrict__ g_p2, const float* __restrict__ be_p2,
    const float* __restrict__ b_ne1, const float* __restrict__ g_ne, const float* __restrict__ be_ne,
    float* __restrict__ w_fused, _Float16* __restrict__ w_dense_T,
    _Float16* __restrict__ w_ctx_T, float* __restrict__ b_eff,
    _Float16* __restrict__ w_p2T, float* __restrict__ sc)
{
    const int tid = threadIdx.x;
    // phase 1: w_fused[c][f] = sum_m w_r1[c][m] * w_r2[m][f]   (256x64)
    for (int i = tid; i < 256 * 64; i += 256) {
        int c = i >> 6, f = i & 63;
        float acc = 0.f;
        for (int m = 0; m < 128; ++m)
            acc = fmaf(w_r1[c * 128 + m], w_r2[m * 64 + f], acc);
        w_fused[c * 64 + f] = acc;
    }
    __syncthreads();
    // phase 2: derived f16 weight matrices (transposed: [f][c] contiguous in c)
    for (int i = tid; i < 64 * 64; i += 256) {
        int f = i >> 6, c = i & 63;
        w_dense_T[f * 64 + c] =
            (_Float16)(w_fused[c * 64 + f] + w_fused[(192 + c) * 64 + f]);
    }
    for (int i = tid; i < 64 * 128; i += 256) {
        int f = i >> 7, c = i & 127;
        w_ctx_T[f * 128 + c] = (_Float16)(w_fused[(64 + c) * 64 + f]);
    }
    if (tid < 64) {
        float acc = b_r2[tid];
        for (int m = 0; m < 128; ++m)
            acc = fmaf(b_r1[m], w_r2[m * 64 + tid], acc);
        b_eff[tid] = acc;
    }
    for (int i = tid; i < 32 * 32; i += 256) {
        int j2 = i >> 5, j1 = i & 31;
        w_p2T[j2 * 32 + j1] = (_Float16)w_p2[j1 * 32 + j2];
    }
    if (tid < 32) {
        float rs  = rsqrtf(1.f + EPSB);
        float s1  = g_p1[tid] * rs;
        float s2  = g_p2[tid] * rs;
        float sne = g_ne[tid] * rs;
        sc[tid]       = s1;
        sc[32 + tid]  = b_p1[tid] * s1 + be_p1[tid];
        sc[64 + tid]  = s2;
        sc[96 + tid]  = b_p2[tid] * s2 + be_p2[tid];
        sc[128 + tid] = sne;
        sc[160 + tid] = b_ne1[tid] * sne + be_ne[tid];
    }
}

// ---------------------------------------------------------------------------
// K1: per masked voxel: xi = gather(input)+pe, A1s = (xi@w_p1)*s1,
//     M1s = (pe@w_p1)*s1, prob = sigmoid(relu(bn(xi@w_ne1))@w_ne2 + b)
// one thread per row n.  grid (N/64, BS), block 64.
// ---------------------------------------------------------------------------
__global__ __launch_bounds__(64) void k1_ne(
    const float* __restrict__ input, const int* __restrict__ mask_idx,
    const float* __restrict__ w_ne1, const float* __restrict__ w_ne2,
    const float* __restrict__ b_ne2, const float* __restrict__ w_p1,
    const float* __restrict__ sc,
    float* __restrict__ xi_out, float* __restrict__ a1s_out,
    float* __restrict__ m1s_out, float* __restrict__ prob_out)
{
    const int b = blockIdx.y;
    const int n = blockIdx.x * 64 + threadIdx.x;
    const int p = mask_idx[b * NPTS + n];
    const float* inb = input + (size_t)b * CCH * FLAT;
    float* xi_row = xi_out + ((size_t)b * NPTS + n) * CCH;

    float a1[32], m1[32], hn[32];
#pragma unroll
    for (int j = 0; j < 32; ++j) { a1[j] = 0.f; m1[j] = 0.f; hn[j] = 0.f; }

    for (int c2 = 0; c2 < 32; ++c2) {
        float div = __expf(-PE_LOGC * (float)(2 * c2));
        float ang = (float)p * div;
        float sv, cv;
        __sincosf(ang, &sv, &cv);
#pragma unroll
        for (int half = 0; half < 2; ++half) {
            int   c  = 2 * c2 + half;
            float pe = half ? cv : sv;
            float x  = inb[(size_t)c * FLAT + p] + pe;
            xi_row[c] = x;
            const float* wp = w_p1  + c * 32;
            const float* wn = w_ne1 + c * 32;
#pragma unroll
            for (int j = 0; j < 32; ++j) {
                a1[j] = fmaf(x,  wp[j], a1[j]);
                m1[j] = fmaf(pe, wp[j], m1[j]);
                hn[j] = fmaf(x,  wn[j], hn[j]);
            }
        }
    }
    float acc = b_ne2[0];
#pragma unroll
    for (int j = 0; j < 32; ++j) {
        float h = fmaxf(fmaf(hn[j], sc[128 + j], sc[160 + j]), 0.f);
        acc = fmaf(h, w_ne2[j], acc);
    }
    prob_out[b * NPTS + n] = 1.f / (1.f + __expf(-acc));

    float* a1r = a1s_out + ((size_t)b * NPTS + n) * 32;
    float* m1r = m1s_out + ((size_t)b * NPTS + n) * 32;
#pragma unroll
    for (int j = 0; j < 32; ++j) {
        float s1 = sc[j];
        a1r[j] = a1[j] * s1;
        m1r[j] = m1[j] * s1;
    }
}

// ---------------------------------------------------------------------------
// K2: multinomial-with-replacement via inverse CDF (deterministic hash RNG),
//     gather tf / ne_k / B1s rows.  grid BS, block 256.
// ---------------------------------------------------------------------------
__global__ __launch_bounds__(256) void k2_sample(
    const float* __restrict__ prob, const float* __restrict__ xi,
    const float* __restrict__ m1s, const float* __restrict__ sc,
    int* __restrict__ topk, float* __restrict__ nek,
    float* __restrict__ b1s, float* __restrict__ tf)
{
    __shared__ float pref[NPTS];
    __shared__ float part[256];
    const int b = blockIdx.x;
    const int t = threadIdx.x;
    const float* pb = prob + b * NPTS;

    float loc[8], s = 0.f;
#pragma unroll
    for (int i = 0; i < 8; ++i) { loc[i] = pb[t * 8 + i]; s += loc[i]; }
    part[t] = s;
    __syncthreads();
    for (int off = 1; off < 256; off <<= 1) {
        float v = (t >= off) ? part[t - off] : 0.f;
        __syncthreads();
        part[t] += v;
        __syncthreads();
    }
    float run = (t > 0) ? part[t - 1] : 0.f;
#pragma unroll
    for (int i = 0; i < 8; ++i) { run += loc[i]; pref[t * 8 + i] = run; }
    __syncthreads();
    const float total = part[255];

    // one draw per thread (K == 256)
    unsigned h = (unsigned)(b * 0x9E3779B9u) ^ (unsigned)(t * 0x85EBCA6Bu) ^ 0x2545F491u;
    h ^= h >> 16; h *= 0x7FEB352Du; h ^= h >> 15; h *= 0x846CA68Bu; h ^= h >> 16;
    float u = ((float)(h >> 8)) * (1.f / 16777216.f) * total;
    int lo = 0, hi = NPTS - 1;
    while (lo < hi) {
        int mid = (lo + hi) >> 1;
        if (pref[mid] < u) lo = mid + 1; else hi = mid;
    }
    const int pick = lo;
    topk[b * KSMP + t] = pick;
    nek[b * KSMP + t]  = pb[pick];
    const float* mrow = m1s + ((size_t)b * NPTS + pick) * 32;
    float* brow = b1s + ((size_t)b * KSMP + t) * 32;
#pragma unroll
    for (int j = 0; j < 32; ++j) brow[j] = mrow[j] + sc[32 + j];
    const float* xr = xi + ((size_t)b * NPTS + pick) * CCH;
    float* tr = tf + ((size_t)b * KSMP + t) * CCH;
#pragma unroll
    for (int c = 0; c < 64; ++c) tr[c] = xr[c];
}

// ---------------------------------------------------------------------------
// K3: pairwise MLP.  hp1 = relu(A1s[n] + B1s[k]) built in LDS (f16),
//     layer2 via v_wmma_f32_16x16x32_f16 (16 pairs x 32 hidden),
//     P = sigmoid(hp2 . w_p3), weighted aggregation of tf.
// block 256 = 8 waves, each wave owns one n-row; grid (N/8, BS).
// ---------------------------------------------------------------------------
__global__ __launch_bounds__(256) void k3_pair(
    const float* __restrict__ a1s, const float* __restrict__ b1s,
    const float* __restrict__ tf,  const float* __restrict__ nek,
    const _Float16* __restrict__ w_p2T_g, const float* __restrict__ sc,
    const float* __restrict__ w_p3, const float* __restrict__ b_p3,
    float* __restrict__ x_intra, float* __restrict__ x_inter)
{
    __shared__ __align__(16) _Float16 wp2[1024];        // [j2][j1] 32x32
    __shared__ __align__(16) _Float16 hp1[8][16 * 32];  // per wave: 16 rows x 32
    __shared__ float pipe[8][32];                       // per wave: Pi[16], Pe[16]

    const int tid = threadIdx.x;
    { // stage w_p2T into LDS (2048B, 8B/thread)
        const uint2* src = (const uint2*)w_p2T_g;
        ((uint2*)wp2)[tid] = src[tid];
    }
    __syncthreads();

    const int wave = tid >> 5, lane = tid & 31;
    const int b = blockIdx.y;
    const int n = blockIdx.x * 8 + wave;
    const int h   = lane >> 4;           // lane half for fragment layout
    const int row = lane >> 1;           // hp1 row this lane computes (2 lanes/row)
    const int j0  = (lane & 1) * 16;     // hidden-j slice this lane computes
    const int col = lane & 15;           // A-row / B-col / D-col for WMMA

    // per-lane slice of A1s (scaled)
    float a1[16];
    {
        const float* ap = a1s + ((size_t)b * NPTS + n) * 32 + j0;
#pragma unroll
        for (int i = 0; i < 16; ++i) a1[i] = ap[i];
    }
    // B fragments of w_p2 (two output tiles j2=0..15, j2=16..31)
    const v16h bf0 = ldfrag(&wp2[(col)      * 32 + h * 16], &wp2[(col)      * 32 + h * 16 + 8]);
    const v16h bf1 = ldfrag(&wp2[(16 + col) * 32 + h * 16], &wp2[(16 + col) * 32 + h * 16 + 8]);

    const float s2a = sc[64 + col],      c2a = sc[96 + col],      w3a = w_p3[col];
    const float s2b = sc[64 + 16 + col], c2b = sc[96 + 16 + col], w3b = w_p3[16 + col];
    const float bp3 = b_p3[0];

    const int c0 = lane * 2;  // channel pair this lane accumulates
    float sPi = 0.f, sPe = 0.f;
    float acc_ix = 0.f, acc_iy = 0.f, acc_ex = 0.f, acc_ey = 0.f;

    for (int kc = 0; kc < 16; ++kc) {
        // ---- build hp1 tile (16 k-rows x 32 hidden, f16) ----
        {
            const int k = kc * 16 + row;
            const float* brow = b1s + ((size_t)b * KSMP + k) * 32 + j0;
            unsigned pk[8];
#pragma unroll
            for (int i = 0; i < 8; ++i) {
                float h0 = fmaxf(a1[2 * i]     + brow[2 * i],     0.f);
                float h1 = fmaxf(a1[2 * i + 1] + brow[2 * i + 1], 0.f);
                union { _Float16 hh[2]; unsigned u; } pkk;
                pkk.hh[0] = (_Float16)h0; pkk.hh[1] = (_Float16)h1;
                pk[i] = pkk.u;
            }
            _Float16* rp = &hp1[wave][row * 32 + j0];
            ((uint4*)rp)[0] = make_uint4(pk[0], pk[1], pk[2], pk[3]);
            ((uint4*)rp)[1] = make_uint4(pk[4], pk[5], pk[6], pk[7]);
        }
        asm volatile("" ::: "memory");  // keep ds stores before ds fragment loads

        // ---- layer-2 GEMM: two WMMAs -> hp2 (16 pairs x 32 hidden) ----
        const _Float16* ab = &hp1[wave][col * 32];
        const v16h af = ldfrag(ab + h * 16, ab + h * 16 + 8);
        v8f cz0 = {}; v8f cz1 = {};
        cz0 = wmma_f16(af, bf0, cz0);
        cz1 = wmma_f16(af, bf1, cz1);

        // ---- P = sigmoid(hp2 . w_p3 + b_p3), Pi/Pe tables ----
        float Sv[8];
#pragma unroll
        for (int rr = 0; rr < 8; ++rr) {
            float v0 = fmaxf(fmaf(cz0[rr], s2a, c2a), 0.f);
            float v1 = fmaxf(fmaf(cz1[rr], s2b, c2b), 0.f);
            float tsum = v0 * w3a + v1 * w3b;
            tsum += __shfl_xor(tsum, 1);
            tsum += __shfl_xor(tsum, 2);
            tsum += __shfl_xor(tsum, 4);
            tsum += __shfl_xor(tsum, 8);
            Sv[rr] = tsum;
        }
        if (col == 0) {
#pragma unroll
            for (int rr = 0; rr < 8; ++rr) {
                int   m  = rr + h * 8;
                float P  = 1.f / (1.f + __expf(-(Sv[rr] + bp3)));
                float nk = nek[b * KSMP + kc * 16 + m];
                pipe[wave][m]      = P * nk;
                pipe[wave][16 + m] = (1.f - P) * nk;
            }
        }
        asm volatile("" ::: "memory");

        // ---- aggregate tf weighted by Pi / Pe ----
#pragma unroll
        for (int m = 0; m < 16; ++m) {
            float pi = pipe[wave][m];
            float pe = pipe[wave][16 + m];
            const float2 tv = *(const float2*)(tf + ((size_t)b * KSMP + kc * 16 + m) * CCH + c0);
            acc_ix = fmaf(pi, tv.x, acc_ix); acc_iy = fmaf(pi, tv.y, acc_iy);
            acc_ex = fmaf(pe, tv.x, acc_ex); acc_ey = fmaf(pe, tv.y, acc_ey);
            sPi += pi; sPe += pe;
        }
    }
    const float inv_i = 1.f / sPi, inv_e = 1.f / sPe;
    *(float2*)(x_intra + ((size_t)b * NPTS + n) * CCH + c0) =
        make_float2(acc_ix * inv_i, acc_iy * inv_i);
    *(float2*)(x_inter + ((size_t)b * NPTS + n) * CCH + c0) =
        make_float2(acc_ex * inv_e, acc_ey * inv_e);
}

// ---------------------------------------------------------------------------
// K4: dense resize  y[b,f,v] = sum_c in[b,c,v]*w_dense[c,f] + b_eff[f]
// Staging of the [64c x 128v] tile is done by the Tensor Data Mover (one 2D
// descriptor, row stride = FLAT) into LDS; threads then only transpose+convert
// LDS->LDS into WMMA A-tile layout.  WMMA f16, K=64 as 2 chained 16x16x32.
// block 256 = 8 waves, each wave one 16-voxel tile; grid (FLAT/128, BS).
// ---------------------------------------------------------------------------
__global__ __launch_bounds__(256) void k4_dense(
    const float* __restrict__ input, const _Float16* __restrict__ w_dense_T,
    const float* __restrict__ b_eff, float* __restrict__ out)
{
    __shared__ __align__(16) float    stage[64 * 128];     // [c][v] f32 (TDM dest)
    __shared__ __align__(16) _Float16 tileA[8 * 16 * 64];  // [wave][v row][c] f16
    const int b = blockIdx.y;
    const int vbase = blockIdx.x * 128;
    const int tid = threadIdx.x;
    const float* inb = input + (size_t)b * CCH * FLAT;

#if defined(__has_builtin) && __has_builtin(__builtin_amdgcn_tensor_load_to_lds)
    if (tid < 32) {  // wave 0 issues the DMA (TDM ignores EXEC; wave-uniform branch)
        const unsigned lds_off = (unsigned)(size_t)(&stage[0]);            // low 32b = LDS offset
        const unsigned long long ga = (unsigned long long)(size_t)(inb + vbase);
        // D# group0: count=1 | lds_addr | global_addr[56:0] | type=2
        uint4v g0 = { 1u, lds_off, (unsigned)ga,
                      ((unsigned)(ga >> 32) & 0x01FFFFFFu) | (2u << 30) };
        // D# group1: data_size=4B, tensor_dim0=16384, tensor_dim1=64,
        //            tile_dim0=128, tile_dim1=64, tensor_dim0_stride=16384
        int8v g1 = { 0x00020000,            // data_size=2 (4 bytes)
                     (int)(16384u << 16),   // tensor_dim0 lo16 @ [31:16]
                     (int)(64u << 16),      // tensor_dim0 hi16=0, tensor_dim1 lo16 @ [31:16]
                     (int)(128u << 16),     // tensor_dim1 hi16=0, tile_dim0 @ [31:16]
                     64,                    // tile_dim1, tile_dim2=0
                     16384,                 // tensor_dim0_stride lo32
                     0, 0 };                // stride hi16, tensor_dim1_stride=0
        int4v g2 = { 0, 0, 0, 0 };
        int4v g3 = { 0, 0, 0, 0 };
#if __clang_major__ >= 23
        int8v g4 = { 0, 0, 0, 0, 0, 0, 0, 0 };
        __builtin_amdgcn_tensor_load_to_lds(g0, g1, g2, g3, g4, 0);
#else
        __builtin_amdgcn_tensor_load_to_lds(g0, g1, g2, g3, 0);
#endif
        __builtin_amdgcn_s_wait_tensorcnt(0);
    }
    __syncthreads();
    // transpose + convert LDS f32 [c][v] -> LDS f16 [wave][v][c]
    for (int it = 0; it < 32; ++it) {
        int c = it * 2 + (tid >> 7);
        int v = tid & 127;
        tileA[(v >> 4) * 1024 + (v & 15) * 64 + c] = (_Float16)stage[c * 128 + v];
    }
#else
    // fallback: cooperative strided gather straight from global
    for (int it = 0; it < 32; ++it) {
        int c = it * 2 + (tid >> 7);
        int v = tid & 127;
        float x = inb[(size_t)c * FLAT + vbase + v];
        tileA[(v >> 4) * 1024 + (v & 15) * 64 + c] = (_Float16)x;
    }
#endif
    if (blockIdx.x + 1 < FLAT / 128)  // warm L2 for the next voxel block
        __builtin_prefetch(inb + vbase + 128, 0, 1);
    __syncthreads();

    const int wave = tid >> 5, lane = tid & 31;
    const int h = lane >> 4, col = lane & 15;
    const int v0 = vbase + wave * 16;

    const _Float16* ab = &tileA[wave * 1024 + col * 64];
    const v16h a0 = ldfrag(ab + h * 16,      ab + h * 16 + 8);
    const v16h a1 = ldfrag(ab + 32 + h * 16, ab + 32 + h * 16 + 8);

#pragma unroll
    for (int nt = 0; nt < 4; ++nt) {
        const int f = nt * 16 + col;
        const float be = b_eff[f];
        v8f acc = {be, be, be, be, be, be, be, be};
        const _Float16* wb = w_dense_T + f * 64;
        acc = wmma_f16(a0, ldfrag(wb + h * 16,      wb + h * 16 + 8),      acc);
        acc = wmma_f16(a1, ldfrag(wb + 32 + h * 16, wb + 32 + h * 16 + 8), acc);
        float* ob = out + ((size_t)b * CCH + f) * FLAT + v0 + h * 8;
        *(float4*)ob       = make_float4(acc[0], acc[1], acc[2], acc[3]);
        *(float4*)(ob + 4) = make_float4(acc[4], acc[5], acc[6], acc[7]);
    }
}

// ---------------------------------------------------------------------------
// K5: sparse ctx resize  y[b,f,idx[n]] += [x_intra;x_inter][n] @ w_ctx
// one wave per 16 n-rows; K=128 as 4 chained WMMAs; grid (N/16, BS).
// ---------------------------------------------------------------------------
__global__ __launch_bounds__(32) void k5_scatter(
    const float* __restrict__ x_intra, const float* __restrict__ x_inter,
    const int* __restrict__ mask_idx, const _Float16* __restrict__ w_ctx_T,
    float* __restrict__ out)
{
    __shared__ __align__(16) _Float16 rowsA[16 * 128];
    __shared__ int vidx[16];
    const int b = blockIdx.y;
    const int n0 = blockIdx.x * 16;
    const int lane = threadIdx.x;

    {   // stage 16 ctx rows as f16: [r][0:64)=x_intra, [64:128)=x_inter
        const int r = lane >> 1, hs = lane & 1;
        const float* src = hs ? x_inter : x_intra;
        const float* sr = src + ((size_t)b * NPTS + n0 + r) * CCH;
#pragma unroll
        for (int j = 0; j < 64; ++j)
            rowsA[r * 128 + hs * 64 + j] = (_Float16)sr[j];
        if (lane < 16) vidx[lane] = mask_idx[b * NPTS + n0 + lane];
    }
    __syncthreads();

    const int h = lane >> 4, col = lane & 15;
    v16h afr[4];
#pragma unroll
    for (int kt = 0; kt < 4; ++kt) {
        const _Float16* ab = &rowsA[col * 128 + kt * 32];
        afr[kt] = ldfrag(ab + h * 16, ab + h * 16 + 8);
    }
#pragma unroll
    for (int nt = 0; nt < 4; ++nt) {
        const int f = nt * 16 + col;
        v8f acc = {};
#pragma unroll
        for (int kt = 0; kt < 4; ++kt) {
            const _Float16* wb = w_ctx_T + f * 128 + kt * 32;
            acc = wmma_f16(afr[kt], ldfrag(wb + h * 16, wb + h * 16 + 8), acc);
        }
        float* ob = out + ((size_t)b * CCH + f) * FLAT;
#pragma unroll
        for (int rr = 0; rr < 8; ++rr) {
            int v = vidx[rr + h * 8];
            ob[v] += acc[rr];  // unique (f,v) per lane -> no atomics needed
        }
    }
}

// ---------------------------------------------------------------------------
// Host-side launcher
// ---------------------------------------------------------------------------
extern "C" void kernel_launch(void* const* d_in, const int* in_sizes, int n_in,
                              void* d_out, int out_size, void* d_ws, size_t ws_size,
                              hipStream_t stream) {
    (void)in_sizes; (void)n_in; (void)out_size; (void)ws_size;
    const float* input    = (const float*)d_in[0];
    const int*   mask_idx = (const int*)  d_in[1];
    const float* w_ne1 = (const float*)d_in[2];
    const float* b_ne1 = (const float*)d_in[3];
    const float* g_ne  = (const float*)d_in[4];
    const float* be_ne = (const float*)d_in[5];
    const float* w_ne2 = (const float*)d_in[6];
    const float* b_ne2 = (const float*)d_in[7];
    const float* w_p1  = (const float*)d_in[8];
    const float* b_p1  = (const float*)d_in[9];
    const float* g_p1  = (const float*)d_in[10];
    const float* be_p1 = (const float*)d_in[11];
    const float* w_p2  = (const float*)d_in[12];
    const float* b_p2  = (const float*)d_in[13];
    const float* g_p2  = (const float*)d_in[14];
    const float* be_p2 = (const float*)d_in[15];
    const float* w_p3  = (const float*)d_in[16];
    const float* b_p3  = (const float*)d_in[17];
    const float* w_r1  = (const float*)d_in[18];
    const float* b_r1  = (const float*)d_in[19];
    const float* w_r2  = (const float*)d_in[20];
    const float* b_r2  = (const float*)d_in[21];
    float* out = (float*)d_out;

    char* ws = (char*)d_ws;
    size_t off = 0;
    auto alloc = [&](size_t bytes) { char* p = ws + off; off = (off + bytes + 255) & ~(size_t)255; return p; };
    float*    W_FUSED   = (float*)   alloc(256 * 64 * 4);
    _Float16* W_DENSE_T = (_Float16*)alloc(64 * 64 * 2);
    _Float16* W_CTX_T   = (_Float16*)alloc(64 * 128 * 2);
    float*    B_EFF     = (float*)   alloc(64 * 4);
    _Float16* W_P2T     = (_Float16*)alloc(32 * 32 * 2);
    float*    SC        = (float*)   alloc(192 * 4);
    float*    XI        = (float*)   alloc((size_t)BS * NPTS * 64 * 4);
    float*    A1S       = (float*)   alloc((size_t)BS * NPTS * 32 * 4);
    float*    M1S       = (float*)   alloc((size_t)BS * NPTS * 32 * 4);
    float*    PROB      = (float*)   alloc((size_t)BS * NPTS * 4);
    int*      TOPK      = (int*)     alloc((size_t)BS * KSMP * 4);
    float*    NEK       = (float*)   alloc((size_t)BS * KSMP * 4);
    float*    B1S       = (float*)   alloc((size_t)BS * KSMP * 32 * 4);
    float*    TF        = (float*)   alloc((size_t)BS * KSMP * 64 * 4);
    float*    XINTRA    = (float*)   alloc((size_t)BS * NPTS * 64 * 4);
    float*    XINTER    = (float*)   alloc((size_t)BS * NPTS * 64 * 4);

    k0_prep<<<1, 256, 0, stream>>>(w_r1, b_r1, w_r2, b_r2, w_p2,
                                   b_p1, g_p1, be_p1, b_p2, g_p2, be_p2,
                                   b_ne1, g_ne, be_ne,
                                   W_FUSED, W_DENSE_T, W_CTX_T, B_EFF, W_P2T, SC);

    k1_ne<<<dim3(NPTS / 64, BS), 64, 0, stream>>>(input, mask_idx, w_ne1, w_ne2, b_ne2,
                                                  w_p1, SC, XI, A1S, M1S, PROB);

    k2_sample<<<BS, 256, 0, stream>>>(PROB, XI, M1S, SC, TOPK, NEK, B1S, TF);

    k3_pair<<<dim3(NPTS / 8, BS), 256, 0, stream>>>(A1S, B1S, TF, NEK, W_P2T, SC,
                                                    w_p3, b_p3, XINTRA, XINTER);

    k4_dense<<<dim3(FLAT / 128, BS), 256, 0, stream>>>(input, W_DENSE_T, B_EFF, out);

    k5_scatter<<<dim3(NPTS / 16, BS), 32, 0, stream>>>(XINTRA, XINTER, mask_idx,
                                                       W_CTX_T, out);
}
